// LSTM_32598801776887
// MI455X (gfx1250) — matrix operand order repaired and checked by reading, not compiled, not run on hardware
//
#include <hip/hip_runtime.h>
#include <math.h>

typedef _Float16 half_t;
typedef __attribute__((ext_vector_type(16))) _Float16 v16h;
typedef __attribute__((ext_vector_type(8)))  float    v8f;
typedef __attribute__((ext_vector_type(2)))  _Float16 half2_t;

// ---------------------------------------------------------------------------
// WMMA fragment loaders (wave32; contiguous k order k = 16*(lane/16) + s, same in A and B)
// ---------------------------------------------------------------------------
// A-matrix 16x32 f16: lane l<16 -> row M=l, holds K = 0..15;
// lane l>=16 -> row M=l-16, holds K = 16..31. 2 halves per VGPR.
__device__ __forceinline__ v16h load_a_frag(const half_t* lds, int row0,
                                            int stride, int kbase, int lane) {
  const int r    = row0 + (lane & 15);
  const int koff = (lane >> 4) ? 16 : 0;
  const half_t* p = lds + r * stride + kbase;
  v16h a;
#pragma unroll
  for (int v = 0; v < 8; ++v) {
    const int k = 2 * v + koff;
    half2_t e = *(const half2_t*)(p + k);   // 4B aligned (stride even, k even)
    a[2 * v]     = e.x;
    a[2 * v + 1] = e.y;
  }
  return a;
}

// B-matrix 32x16 f16 stored n-major in LDS (Bs[n][k]): lane l<16 -> col N=l,
// K=0..15 ; lane l>=16 -> col N=l-16, K=16..31. 2 consecutive K per VGPR.
__device__ __forceinline__ v16h load_b_frag(const half_t* lds, int ncol0,
                                            int stride, int lane) {
  const int n    = ncol0 + (lane & 15);
  const int koff = (lane >> 4) ? 16 : 0;
  const half_t* p = lds + n * stride + koff;
  v16h b;
#pragma unroll
  for (int v = 0; v < 8; ++v) {
    half2_t e = *(const half2_t*)(p + 2 * v);
    b[2 * v]     = e.x;
    b[2 * v + 1] = e.y;
  }
  return b;
}

// convert 8 f32 -> 8 f16 and store to LDS as 4x half2
__device__ __forceinline__ void cvt_store8(half_t* d, float4 f0, float4 f1) {
  half2_t p;
  p.x = (half_t)f0.x; p.y = (half_t)f0.y; *(half2_t*)(d + 0) = p;
  p.x = (half_t)f0.z; p.y = (half_t)f0.w; *(half2_t*)(d + 2) = p;
  p.x = (half_t)f1.x; p.y = (half_t)f1.y; *(half2_t*)(d + 4) = p;
  p.x = (half_t)f1.z; p.y = (half_t)f1.w; *(half2_t*)(d + 6) = p;
}

// ---------------------------------------------------------------------------
// Kernel 1: x_proj[t*256+b, 0..399] = x[b,t,:] . W_ih^T + (b_ih + b_hh)
// M = 65536 (b*256+t), N = 400, K = 2016. 512 blocks x 512 threads.
// Software-pipelined twice over:
//   * global loads for K-step k+1 issue before the WMMA phase of step k
//   * B-fragment DS loads for N-tile ti+1 issue before WMMA of tile ti
// ---------------------------------------------------------------------------
#define A_STRIDE 34   // padded row stride (halves) -> conflict-free ds reads
#define B_STRIDE 34

__global__ __launch_bounds__(512, 1)
void lstm_xproj_gemm(const float* __restrict__ x,
                     const float* __restrict__ W_ih,
                     const float* __restrict__ b_ih,
                     const float* __restrict__ b_hh,
                     float* __restrict__ xproj) {
  __shared__ half_t As[128 * A_STRIDE];   //  8.7 KB
  __shared__ half_t Bs[400 * B_STRIDE];   // 27.2 KB

  const int tid   = threadIdx.x;
  const int lane  = tid & 31;
  const int wave  = tid >> 5;          // 0..15
  const int m0    = blockIdx.x * 128;
  const int mtile = wave >> 1;         // 0..7 : M-tile within block
  const int nhalf = wave & 1;          // split 25 N-tiles 13/12 per wave pair
  const int nt0   = nhalf ? 13 : 0;
  const int ncnt  = nhalf ? 12 : 13;

  // staging coordinates: 8 contiguous f32 per thread per matrix row-group
  const int sr = tid >> 2;             // 0..127
  const int sc = (tid & 3) * 8;        // 0,8,16,24

  v8f acc[13];
#pragma unroll
  for (int i = 0; i < 13; ++i)
#pragma unroll
    for (int j = 0; j < 8; ++j) acc[i][j] = 0.0f;

  auto load_tile = [&](int k0, float4 (&ta)[2], float4 (&tb)[4][2]) {
    const float* srcA = x + (size_t)(m0 + sr) * 2016 + k0 + sc;
    ta[0] = *(const float4*)(srcA);
    ta[1] = *(const float4*)(srcA + 4);
#pragma unroll
    for (int i = 0; i < 4; ++i) {
      const int n = sr + i * 128;
      if (n < 400) {
        const float* srcB = W_ih + (size_t)n * 2016 + k0 + sc;
        tb[i][0] = *(const float4*)(srcB);
        tb[i][1] = *(const float4*)(srcB + 4);
      }
    }
  };

  auto store_tile = [&](const float4 (&ta)[2], const float4 (&tb)[4][2]) {
    cvt_store8(As + sr * A_STRIDE + sc, ta[0], ta[1]);
#pragma unroll
    for (int i = 0; i < 4; ++i) {
      const int n = sr + i * 128;
      if (n < 400) cvt_store8(Bs + n * B_STRIDE + sc, tb[i][0], tb[i][1]);
    }
  };

  auto gemm_step = [&](int ki, const float4 (&ca)[2], const float4 (&cb)[4][2],
                       float4 (&na)[2], float4 (&nb)[4][2]) {
    __syncthreads();                        // all waves done reading LDS
    store_tile(ca, cb);
    if (ki + 1 < 63) load_tile((ki + 1) * 32, na, nb);  // in flight during WMMA
    __syncthreads();                        // staged tile visible
    const v16h afrag = load_a_frag(As, mtile * 16, A_STRIDE, 0, lane);
    // B-fragment double buffering: tile ti+1 loads issue before WMMA of ti,
    // so each WMMA waits only on loads issued one full WMMA earlier.
    v16h bcur = load_b_frag(Bs, nt0 * 16, B_STRIDE, lane);
#pragma unroll
    for (int ti = 0; ti < 13; ++ti) {
      if (ti < ncnt) {   // wave-uniform guard: EXEC stays all-ones
        v16h bnext = bcur;
        if (ti + 1 < ncnt)
          bnext = load_b_frag(Bs, (nt0 + ti + 1) * 16, B_STRIDE, lane);
        acc[ti] = __builtin_amdgcn_wmma_f32_16x16x32_f16(
            false, afrag, false, bcur, (short)0, acc[ti], false, false);
        bcur = bnext;
      }
    }
  };

  // two named register staging sets, ping-ponged (no dynamic indexing)
  float4 arA[2], brA[4][2], arB[2], brB[4][2];
  load_tile(0, arA, brA);
#pragma unroll 1
  for (int ki = 0; ki < 63; ki += 2) {
    gemm_step(ki, arA, brA, arB, brB);
    if (ki + 1 < 63) gemm_step(ki + 1, arB, brB, arA, brA);
  }

  // --- epilogue: add bias, store permuted to [t, b, 400] layout ---
  const int nl = lane & 15;
  const int hi = lane >> 4;
#pragma unroll
  for (int ti = 0; ti < 13; ++ti) {
    if (ti < ncnt) {
      const int n = (nt0 + ti) * 16 + nl;
      const float bias = b_ih[n] + b_hh[n];
#pragma unroll
      for (int d = 0; d < 8; ++d) {
        const int m  = m0 + mtile * 16 + d + hi * 8;  // = b*256 + t
        const int b  = m >> 8;
        const int tt = m & 255;
        xproj[(size_t)(tt * 256 + b) * 400 + n] = acc[ti][d] + bias;
      }
    }
  }
}

// ---------------------------------------------------------------------------
// Kernel 2: recurrent scan + fused FC. 16 blocks x 256 threads (8 waves).
// Block handles 16 batch rows for all 256 timesteps (batch-independent scan).
// W_hh lives in registers as B-fragments for the entire scan.
// ---------------------------------------------------------------------------
#define HSTR 134   // h-tile LDS row stride in halves (16 x 128, zero padded)
#define GSTR 404   // gates LDS row stride in floats

__device__ __forceinline__ float fast_sigmoid(float xv) {
  return __builtin_amdgcn_rcpf(1.0f + __expf(-xv));
}

__global__ __launch_bounds__(256, 1)
void lstm_scan(const float* __restrict__ xproj,
               const float* __restrict__ W_hh,
               const float* __restrict__ W_fc,
               const float* __restrict__ b_fc,
               float* __restrict__ out) {
  __shared__ half_t hbuf[16 * HSTR];    // h as f16 A-tile (K padded to 128)
  __shared__ float  gates[16 * GSTR];   // 16 x 400 gate staging
  __shared__ float  hfin[16 * 100];     // final-step h in f32 for the FC

  const int tid  = threadIdx.x;
  const int lane = tid & 31;
  const int wave = tid >> 5;            // 0..7
  const int b0   = blockIdx.x * 16;
  const int ntiles = (wave == 0) ? 4 : 3;   // n-tiles: wave, wave+8, wave+16[,24]

  // --- build W_hh^T B-fragments in registers (loaded from HBM once) ---
  v16h bw[4][4];
  {
    const int nl    = lane & 15;
    const int koff2 = (lane >> 4) ? 16 : 0;
#pragma unroll
    for (int s = 0; s < 4; ++s) {
      const int  nt    = wave + 8 * s;
      const bool valid = (s < 3) || (wave == 0);
      const int  n     = nt * 16 + nl;          // gate index (0..399 if valid)
#pragma unroll
      for (int kf = 0; kf < 4; ++kf) {
        v16h f;
#pragma unroll
        for (int v = 0; v < 8; ++v) {
          const int k = kf * 32 + koff2 + 2 * v;
          const float e0 = (valid && (k     < 100)) ? W_hh[n * 100 + k]     : 0.0f;
          const float e1 = (valid && (k + 1 < 100)) ? W_hh[n * 100 + k + 1] : 0.0f;
          f[2 * v]     = (half_t)e0;
          f[2 * v + 1] = (half_t)e1;
        }
        bw[s][kf] = f;
      }
    }
  }

  float c_reg[7];
#pragma unroll
  for (int i = 0; i < 7; ++i) c_reg[i] = 0.0f;

  // zero h tile (incl. K=100..127 padding — padding stays zero forever)
  for (int i = tid; i < 16 * HSTR; i += 256) hbuf[i] = (half_t)0.0f;
  __syncthreads();

  const int nl = lane & 15;
  const int hi = lane >> 4;

#pragma unroll 1
  for (int t = 0; t < 256; ++t) {
    // ---- issue all x_proj C-fragment loads first (independent of h/LDS) ----
    v8f cf[4];
#pragma unroll
    for (int s = 0; s < 4; ++s) {
      if (s < ntiles) {   // wave-uniform
        const int nt = wave + 8 * s;
        const float* src = xproj + ((size_t)t * 256 + b0) * 400 + nt * 16 + nl;
#pragma unroll
        for (int d = 0; d < 8; ++d) cf[s][d] = src[(size_t)(d + hi * 8) * 400];
      }
    }

    // ---- A-fragments of current h (same 16x128 tile for all waves) ----
    v16h a[4];
#pragma unroll
    for (int kf = 0; kf < 4; ++kf)
      a[kf] = load_a_frag(hbuf, 0, HSTR, kf * 32, lane);

#pragma unroll
    for (int s = 0; s < 4; ++s) {
      if (s < ntiles) {
        const int nt = wave + 8 * s;
        v8f c = cf[s];
#pragma unroll
        for (int kf = 0; kf < 4; ++kf)
          c = __builtin_amdgcn_wmma_f32_16x16x32_f16(
              false, a[kf], false, bw[s][kf], (short)0, c, false, false);
#pragma unroll
        for (int d = 0; d < 8; ++d)
          gates[(d + hi * 8) * GSTR + nt * 16 + nl] = c[d];
      }
    }
    __syncthreads();

    // ---- prefetch next step's x_proj tile (global_prefetch_b8) ----
    if (t + 1 < 256) {
      const char* np = (const char*)(xproj + ((size_t)(t + 1) * 256 + b0) * 400);
      const int off = tid * 128;
      if (off < 16 * 400 * 4) __builtin_prefetch(np + off, 0, 1);
    }

    // ---- pointwise LSTM cell: 1600 (b,j) elements, c-state in registers ----
#pragma unroll
    for (int i = 0; i < 7; ++i) {
      const int idx = tid + i * 256;
      if (idx < 1600) {
        const int b = idx / 100;
        const int j = idx - b * 100;
        const float gi = gates[b * GSTR + j];
        const float gf = gates[b * GSTR + 100 + j];
        const float gg = gates[b * GSTR + 200 + j];
        const float go = gates[b * GSTR + 300 + j];
        const float ig = fast_sigmoid(gi);
        const float fg = fast_sigmoid(gf);
        const float gt = 2.0f * fast_sigmoid(2.0f * gg) - 1.0f;   // tanh
        const float og = fast_sigmoid(go);
        const float c  = fg * c_reg[i] + ig * gt;
        c_reg[i] = c;
        const float h = og * (2.0f * fast_sigmoid(2.0f * c) - 1.0f);
        hbuf[b * HSTR + j] = (half_t)h;
        if (t == 255) hfin[b * 100 + j] = h;
      }
    }
    __syncthreads();
  }

  // fused FC: out[b0+b, k] = h_T[b,:] . W_fc[k,:] + b_fc[k]
  if (tid < 48) {
    const int b = tid / 3;
    const int k = tid - b * 3;
    float s = b_fc[k];
#pragma unroll 4
    for (int j = 0; j < 100; ++j) s += hfin[b * 100 + j] * W_fc[k * 100 + j];
    out[(b0 + b) * 3 + k] = s;
  }
}

// ---------------------------------------------------------------------------
extern "C" void kernel_launch(void* const* d_in, const int* in_sizes, int n_in,
                              void* d_out, int out_size, void* d_ws, size_t ws_size,
                              hipStream_t stream) {
  (void)in_sizes; (void)n_in; (void)out_size; (void)ws_size;
  const float* x    = (const float*)d_in[0];
  const float* W_ih = (const float*)d_in[1];
  const float* W_hh = (const float*)d_in[2];
  const float* b_ih = (const float*)d_in[3];
  const float* b_hh = (const float*)d_in[4];
  const float* W_fc = (const float*)d_in[5];
  const float* b_fc = (const float*)d_in[6];
  float* out   = (float*)d_out;
  float* xproj = (float*)d_ws;   // [256(T), 256(B), 400] f32 = 104.9 MB

  lstm_xproj_gemm<<<512, 512, 0, stream>>>(x, W_ih, b_ih, b_hh, xproj);
  lstm_scan<<<16, 256, 0, stream>>>(xproj, W_hh, W_fc, b_fc, out);
}
